// MyModel_39943195853194
// MI455X (gfx1250) — matrix operand order repaired and checked by reading, not compiled, not run on hardware
//
#include <hip/hip_runtime.h>
#include <hip/hip_bf16.h>

// ---------------------------------------------------------------------------
// GraphSAGE (2 layers, 2 graphs) for MI455X / gfx1250.
//   - edge aggregation: wave-per-edge, L2-resident float atomics
//   - SAGE layer GEMM: v_wmma_f32_16x16x32_bf16, B fragments held in VGPRs,
//     A tile staged through padded LDS (float4 loads, 4x-packed bf16 stores)
//   - layer-2 column means accumulated in registers, single atomic flush
// ---------------------------------------------------------------------------

typedef __attribute__((ext_vector_type(16))) __bf16 v16bf;
typedef __attribute__((ext_vector_type(4)))  __bf16 v4bf;
typedef __attribute__((ext_vector_type(8)))  float  v8f;

// ------------------------- utility kernels ---------------------------------

__global__ void zero_f32(float* __restrict__ p, long n) {
    long i = (long)blockIdx.x * blockDim.x + threadIdx.x;
    if (i < n) p[i] = 0.0f;
}

__global__ void degree_kernel(const int* __restrict__ dst, float* __restrict__ deg,
                              int n_edges) {
    int i = blockIdx.x * blockDim.x + threadIdx.x;
    if (i < n_edges) atomicAdd(&deg[dst[i]], 1.0f);
}

// One wave per edge; lanes stride over channels. x is L2-resident (<=51MB).
template <int D>
__global__ __launch_bounds__(256) void edge_agg(const float* __restrict__ x,
                                                const int* __restrict__ src,
                                                const int* __restrict__ dst,
                                                float* __restrict__ agg,
                                                int n_edges) {
    int e = (blockIdx.x * blockDim.x + threadIdx.x) >> 5;
    int lane = threadIdx.x & 31;
    if (e >= n_edges) return;
    int s = src[e];
    int t = dst[e];
    const float* xs = x + (long)s * D;
    float*       at = agg + (long)t * D;
#pragma unroll
    for (int c = lane; c < D; c += 32)
        atomicAdd(&at[c], xs[c]);
}

// ------------------------- WMMA SAGE layer ----------------------------------
// Computes H = relu([X | agg/deg] @ [Wself ; Wneigh]) for 128 output columns.
// KT = total K (2*KIN). 8 waves per block: wave w owns output cols [16w,16w+16).
// If colsum != nullptr, instead of storing H the kernel accumulates per-column
// sums of relu output (for the graph-mean readout).

template <int KT>
__global__ __launch_bounds__(256) void sage_gemm(
    const float* __restrict__ X,      // [N, KT/2]
    const float* __restrict__ Magg,   // [N, KT/2] (neighbor sums)
    const float* __restrict__ deg,    // [N]
    const float* __restrict__ Wself,  // [KT/2, 128]
    const float* __restrict__ Wneigh, // [KT/2, 128]
    float* __restrict__ Hout,         // [N,128] or nullptr
    float* __restrict__ colsum,       // [128]  or nullptr
    int n_nodes)
{
    constexpr int KIN   = KT / 2;
    constexpr int NSTEP = KT / 32;     // wmma K-steps
    constexpr int ASTR  = KT + 8;      // LDS row stride (bf16) — bank-conflict pad
    constexpr int K4    = KT / 4;      // float4 chunks per row
    constexpr int NCHNK = 16 * K4;     // float4 chunks per 16-row tile

    __shared__ __bf16 A_lds[16 * ASTR];
    __shared__ float  invdeg_s[16];

    const int wave = threadIdx.x >> 5;
    const int lane = threadIdx.x & 31;
    const int hlf  = lane >> 4;        // lane group 0..15 vs 16..31
    const int c    = lane & 15;        // output column within tile / A row
    const int n0   = wave * 16;        // this wave's output column base

    // ---- load this wave's B fragments into registers, keep for whole kernel.
    // B k order used here (contiguous, matched by A): lanes 0-15 hold K=0..15, lanes 16-31 K=16..31,
    // VGPR v packs K = 16*hlf + {2v, 2v+1} for column n0+c.
    v16bf bfrag[NSTEP];
#pragma unroll
    for (int s = 0; s < NSTEP; ++s) {
#pragma unroll
        for (int v = 0; v < 8; ++v) {
            int k0 = s * 32 + hlf * 16 + 2 * v;
            int k1 = k0 + 1;
            const float* w0 = (k0 < KIN) ? (Wself  + (long)k0 * 128)
                                         : (Wneigh + (long)(k0 - KIN) * 128);
            const float* w1 = (k1 < KIN) ? (Wself  + (long)k1 * 128)
                                         : (Wneigh + (long)(k1 - KIN) * 128);
            bfrag[s][2 * v]     = (__bf16)w0[n0 + c];
            bfrag[s][2 * v + 1] = (__bf16)w1[n0 + c];
        }
    }

    v8f csum = {};                     // per-lane running column sums (readout)

    const int n_blocks = n_nodes >> 4; // 16-row blocks (100000/16 = 6250 exact)
    for (int blk = blockIdx.x; blk < n_blocks; blk += gridDim.x) {
        const int row0 = blk << 4;

        __syncthreads();               // previous iteration's readers done
        if (threadIdx.x < 16) {
            float dg = deg[row0 + threadIdx.x];
            invdeg_s[threadIdx.x] = 1.0f / fmaxf(dg, 1.0f);
        }
        __syncthreads();

        // ---- cooperative A-tile load: float4 global loads, 4x bf16 LDS store.
        // KIN is a multiple of 4, so a chunk never straddles the X/Magg seam.
#pragma unroll
        for (int ci = threadIdx.x; ci < NCHNK; ci += 256) {
            int r  = ci / K4;
            int k4 = (ci - r * K4) * 4;
            int node = row0 + r;
            float4 v;
            if (k4 < KIN) {
                v = *(const float4*)(X + (long)node * KIN + k4);
            } else {
                v = *(const float4*)(Magg + (long)node * KIN + (k4 - KIN));
                float idg = invdeg_s[r];
                v.x *= idg; v.y *= idg; v.z *= idg; v.w *= idg;
            }
            v4bf b;
            b[0] = (__bf16)v.x; b[1] = (__bf16)v.y;
            b[2] = (__bf16)v.z; b[3] = (__bf16)v.w;
            *(v4bf*)&A_lds[r * ASTR + k4] = b;
        }
        __syncthreads();

        // ---- WMMA over K. A 16x32 bf16 layout: lane row M = c; element
        // j (0..15) holds K = ks + 16*hlf + j (same order as B fragments).
        v8f acc = {};
#pragma unroll
        for (int s = 0; s < NSTEP; ++s) {
            const __bf16* ap = &A_lds[c * ASTR + s * 32 + hlf * 16];
            v16bf afrag;
#pragma unroll
            for (int j = 0; j < 8; ++j) afrag[j] = ap[j];
#pragma unroll
            for (int j = 0; j < 8; ++j) afrag[8 + j] = ap[8 + j];
            acc = __builtin_amdgcn_wmma_f32_16x16x32_bf16(
                false, afrag, false, bfrag[s], (short)0, acc, false, false);
        }

        // ---- C layout: VGPR r, lanes 0-15 -> row r, lanes 16-31 -> row r+8.
        // Single base address; r*128 folds into the store instruction offset.
        if (Hout) {
            float* base = Hout + ((long)(row0 + hlf * 8) * 128 + n0 + c);
#pragma unroll
            for (int r = 0; r < 8; ++r)
                base[r * 128] = fmaxf(acc[r], 0.0f);
        }
        if (colsum) {
#pragma unroll
            for (int r = 0; r < 8; ++r) csum[r] += fmaxf(acc[r], 0.0f);
        }
    }

    if (colsum) {
        float s = 0.0f;
#pragma unroll
        for (int r = 0; r < 8; ++r) s += csum[r];
        s += __shfl_xor(s, 16, 32);    // combine the two lane halves (same col)
        if (hlf == 0) atomicAdd(&colsum[n0 + c], s);
    }
}

// ------------------------- readout ------------------------------------------

__global__ void readout_kernel(const float* __restrict__ cs1,
                               const float* __restrict__ cs2,
                               const float* __restrict__ Wlin1,
                               const float* __restrict__ Wlin2,
                               float* __restrict__ out, float inv_n) {
    int j = threadIdx.x;
    if (j >= 16) return;
    float s1 = 0.0f, s2 = 0.0f;
    for (int k = 0; k < 128; ++k) {
        s1 += cs1[k] * inv_n * Wlin1[k * 16 + j];
        s2 += cs2[k] * inv_n * Wlin2[k * 16 + j];
    }
    float o1 = 1.0f / (1.0f + __expf(-s1));
    float o2 = 1.0f / (1.0f + __expf(-s2));
    out[j] = 0.5f * (o1 + o2);
}

// ------------------------- launch --------------------------------------------

static inline size_t align256(size_t x) { return (x + 255) & ~(size_t)255; }

static void run_graph(const float* feats, const int* src, const int* dst,
                      const float* Wself1, const float* Wneigh1,
                      const float* Wself2, const float* Wneigh2,
                      float* deg, float* agg, float* h, float* colsum,
                      int N, int E, hipStream_t stream) {
    const int ZT = 256;
    // degree (shared by both layers of this graph)
    zero_f32<<<(N + ZT - 1) / ZT, ZT, 0, stream>>>(deg, N);
    degree_kernel<<<(E + ZT - 1) / ZT, ZT, 0, stream>>>(dst, deg, E);

    // ---- layer 1: aggregate feats (d=64), GEMM K=128 -> h [N,128]
    zero_f32<<<((long)N * 64 + ZT - 1) / ZT, ZT, 0, stream>>>(agg, (long)N * 64);
    edge_agg<64><<<(E * 8 + ZT - 1) / ZT, ZT, 0, stream>>>(feats, src, dst, agg, E);
    sage_gemm<128><<<1024, 256, 0, stream>>>(feats, agg, deg, Wself1, Wneigh1,
                                             h, (float*)nullptr, N);

    // ---- layer 2: aggregate h (d=128), GEMM K=256 -> column sums only
    zero_f32<<<((long)N * 128 + ZT - 1) / ZT, ZT, 0, stream>>>(agg, (long)N * 128);
    edge_agg<128><<<(E * 8 + ZT - 1) / ZT, ZT, 0, stream>>>(h, src, dst, agg, E);
    sage_gemm<256><<<1024, 256, 0, stream>>>(h, agg, deg, Wself2, Wneigh2,
                                             (float*)nullptr, colsum, N);
}

extern "C" void kernel_launch(void* const* d_in, const int* in_sizes, int n_in,
                              void* d_out, int out_size, void* d_ws, size_t ws_size,
                              hipStream_t stream) {
    const float* feats    = (const float*)d_in[0];
    const int*   H1_src   = (const int*)d_in[1];
    const int*   H1_dst   = (const int*)d_in[2];
    const int*   H2_src   = (const int*)d_in[3];
    const int*   H2_dst   = (const int*)d_in[4];
    const float* W1_self1 = (const float*)d_in[5];
    const float* W1_neigh1= (const float*)d_in[6];
    const float* W1_self2 = (const float*)d_in[7];
    const float* W1_neigh2= (const float*)d_in[8];
    const float* W1_lin   = (const float*)d_in[9];
    const float* W2_self1 = (const float*)d_in[10];
    const float* W2_neigh1= (const float*)d_in[11];
    const float* W2_self2 = (const float*)d_in[12];
    const float* W2_neigh2= (const float*)d_in[13];
    const float* W2_lin   = (const float*)d_in[14];

    const int N = in_sizes[0] / 64;   // 100000
    const int E = in_sizes[1];        // 1600000

    // workspace carve-up
    char* ws = (char*)d_ws;
    size_t off = 0;
    float* deg = (float*)(ws + off); off += align256((size_t)N * 4);
    float* agg = (float*)(ws + off); off += align256((size_t)N * 128 * 4);
    float* h   = (float*)(ws + off); off += align256((size_t)N * 128 * 4);
    float* cs1 = (float*)(ws + off); off += align256(128 * 4);
    float* cs2 = (float*)(ws + off); off += align256(128 * 4);
    (void)ws_size; (void)n_in;

    zero_f32<<<1, 256, 0, stream>>>(cs1, 128);
    zero_f32<<<1, 256, 0, stream>>>(cs2, 128);

    run_graph(feats, H1_src, H1_dst, W1_self1, W1_neigh1, W1_self2, W1_neigh2,
              deg, agg, h, cs1, N, E, stream);
    run_graph(feats, H2_src, H2_dst, W2_self1, W2_neigh1, W2_self2, W2_neigh2,
              deg, agg, h, cs2, N, E, stream);

    readout_kernel<<<1, 32, 0, stream>>>(cs1, cs2, W1_lin, W2_lin,
                                         (float*)d_out, 1.0f / (float)N);
}
